// SetAbstraction_70325794505117
// MI455X (gfx1250) — matrix operand order repaired and checked by reading, not compiled, not run on hardware
//
#include <hip/hip_runtime.h>

#define B_   8
#define N_   8192
#define S_   2048
#define K_   32
#define C_   32
#define DIN  35     // 3 + C input channels
#define DK   64     // K-dim of collapsed GEMM, padded 35 -> 64
#define DOUT 128

typedef __attribute__((ext_vector_type(16))) __bf16 v16bf;
typedef __attribute__((ext_vector_type(8)))  float  v8f;

union FragU { v16bf v; float4 f[2]; };

// round-to-nearest-even f32 -> bf16 (manual, avoids relying on __bf16 conversions)
__device__ inline unsigned short f2bf(float x) {
  unsigned u = __float_as_uint(x);
  return (unsigned short)((u + 0x7FFFu + ((u >> 16) & 1u)) >> 16);
}
__device__ inline unsigned pack2(float a, float b) {
  return (unsigned)f2bf(a) | ((unsigned)f2bf(b) << 16);
}

// ---------------------------------------------------------------------------
// Fold the 3 linear layers (no nonlinearity in reference) into one GEMM:
//   Wc[c][n] = (W0@W1@W2)[c][n]  (c padded to 64, stored transposed [n][k] bf16)
//   bc[n]    = (b0@W1 + b1)@W2 + b2
// ---------------------------------------------------------------------------
__global__ void compose_kernel(const float* __restrict__ w0, const float* __restrict__ b0,
                               const float* __restrict__ w1, const float* __restrict__ b1,
                               const float* __restrict__ w2, const float* __restrict__ b2,
                               unsigned short* __restrict__ wct, float* __restrict__ bc) {
  __shared__ float W01[DIN * 64];
  __shared__ float b01[64];
  const int tid = threadIdx.x;
  for (int i = tid; i < DIN * 64; i += 256) {
    int c = i >> 6, e2 = i & 63;
    float s = 0.f;
    for (int e = 0; e < 64; ++e) s += w0[c * 64 + e] * w1[e * 64 + e2];
    W01[i] = s;
  }
  for (int i = tid; i < 64; i += 256) {
    float s = b1[i];
    for (int e = 0; e < 64; ++e) s += b0[e] * w1[e * 64 + i];
    b01[i] = s;
  }
  __syncthreads();
  for (int i = tid; i < DOUT * DK; i += 256) {
    int n = i >> 6, c = i & 63;
    float s = 0.f;
    if (c < DIN)
      for (int e2 = 0; e2 < 64; ++e2) s += W01[c * 64 + e2] * w2[e2 * DOUT + n];
    wct[i] = f2bf(s);                     // transposed layout: [n][k]
  }
  for (int i = tid; i < DOUT; i += 256) {
    float s = b2[i];
    for (int e2 = 0; e2 < 64; ++e2) s += b01[e2] * w2[e2 * DOUT + i];
    bc[i] = s;
  }
}

// ---------------------------------------------------------------------------
// FPS: one workgroup (1024 thr = 32 waves) per batch; cloud in LDS (96KB),
// per-thread coords+dist in registers; 2048 serial argmax steps.
// ---------------------------------------------------------------------------
__global__ __launch_bounds__(1024) void fps_kernel(const float* __restrict__ xyz,
                                                   float* __restrict__ newxyz) {
  extern __shared__ float smem[];
  float* sx = smem;  float* sy = sx + N_;  float* sz = sy + N_;
  float* rv = sz + N_;
  int*   ri = (int*)(rv + 32);
  int*   bcst = ri + 32;
  const int b = blockIdx.x, tid = threadIdx.x;
  const float* base = xyz + (size_t)b * N_ * 3;
  for (int i = tid; i < N_; i += 1024) {
    sx[i] = base[i * 3 + 0]; sy[i] = base[i * 3 + 1]; sz[i] = base[i * 3 + 2];
  }
  __syncthreads();
  float px[8], py[8], pz[8], dist[8];
#pragma unroll
  for (int i = 0; i < 8; ++i) {
    int p = tid + i * 1024;
    px[i] = sx[p]; py[i] = sy[p]; pz[i] = sz[p];
    dist[i] = 1e10f;
  }
  const int lane = tid & 31, wid = tid >> 5;
  int last = 0;
  for (int s = 0; s < S_; ++s) {
    float lx = sx[last], ly = sy[last], lz = sz[last];
    if (tid == 0) {
      float* o = newxyz + ((size_t)b * S_ + s) * 3;
      o[0] = lx; o[1] = ly; o[2] = lz;
    }
    float bv = -1.f; int bi = 0x7fffffff;
#pragma unroll
    for (int i = 0; i < 8; ++i) {
      float dx = px[i] - lx, dy = py[i] - ly, dz = pz[i] - lz;
      float d = dx * dx + dy * dy + dz * dz;
      dist[i] = fminf(dist[i], d);
      int p = tid + i * 1024;
      if (dist[i] > bv || (dist[i] == bv && p < bi)) { bv = dist[i]; bi = p; }
    }
#pragma unroll
    for (int off = 16; off > 0; off >>= 1) {     // wave32 argmax (first-max idx)
      float ov = __shfl_down(bv, off, 32);
      int   oi = __shfl_down(bi, off, 32);
      if (ov > bv || (ov == bv && oi < bi)) { bv = ov; bi = oi; }
    }
    if (lane == 0) { rv[wid] = bv; ri[wid] = bi; }
    __syncthreads();
    if (wid == 0) {
      bv = rv[lane]; bi = ri[lane];
#pragma unroll
      for (int off = 16; off > 0; off >>= 1) {
        float ov = __shfl_down(bv, off, 32);
        int   oi = __shfl_down(bi, off, 32);
        if (ov > bv || (ov == bv && oi < bi)) { bv = ov; bi = oi; }
      }
      if (lane == 0) *bcst = bi;
    }
    __syncthreads();
    last = *bcst;
  }
}

// ---------------------------------------------------------------------------
// Ball query: one wave per centroid, batch cloud in LDS; ordered first-K
// compaction via wave32 ballot + prefix-popcount; early exit at K.
// ---------------------------------------------------------------------------
__global__ __launch_bounds__(256) void ball_kernel(const float* __restrict__ xyz,
                                                   const float* __restrict__ newxyz,
                                                   int* __restrict__ idx_out) {
  extern __shared__ float smem[];
  float* sx = smem;  float* sy = sx + N_;  float* sz = sy + N_;
  const int bpb = S_ / 8;
  const int b = blockIdx.x / bpb;
  const int s0 = (blockIdx.x % bpb) * 8;
  const int tid = threadIdx.x;
  const float* base = xyz + (size_t)b * N_ * 3;
  for (int i = tid; i < N_; i += 256) {
    sx[i] = base[i * 3 + 0]; sy[i] = base[i * 3 + 1]; sz[i] = base[i * 3 + 2];
  }
  __syncthreads();
  const int wid = tid >> 5, lane = tid & 31;
  const size_t row = (size_t)b * S_ + (s0 + wid);
  const float cx = newxyz[row * 3 + 0], cy = newxyz[row * 3 + 1], cz = newxyz[row * 3 + 2];
  int* out = idx_out + row * K_;
  const float r2 = 0.2f * 0.2f;
  int count = 0, first = 0; bool found = false;
  for (int b0 = 0; b0 < N_; b0 += 32) {
    int p = b0 + lane;
    float dx = sx[p] - cx, dy = sy[p] - cy, dz = sz[p] - cz;
    bool w = (dx * dx + dy * dy + dz * dz) < r2;
    unsigned mask = (unsigned)__ballot(w);          // wave32: low 32 bits
    if (!found && mask) { first = b0 + (__ffs(mask) - 1); found = true; }
    int pos = count + __popc(mask & ((1u << lane) - 1u));
    if (w && pos < K_) out[pos] = p;
    count += __popc(mask);
    if (count >= K_) break;                         // uniform across wave
  }
  if (lane >= count) out[lane] = found ? first : 0; // reference padding rule
}

// ---------------------------------------------------------------------------
// Fused gather + collapsed GEMM (WMMA bf16) + maxpool. One wave per centroid:
// A = feat (32 neighbors x 64 padded ch), B = WcT in LDS, D-tiles max-reduced.
// ---------------------------------------------------------------------------
__global__ __launch_bounds__(256) void mlp_kernel(const float* __restrict__ xyz,
                                                  const float* __restrict__ points,
                                                  const unsigned short* __restrict__ wct,
                                                  const float* __restrict__ bcomb,
                                                  const float* __restrict__ newxyz,
                                                  const int* __restrict__ idx,
                                                  float* __restrict__ newpts) {
  __shared__ __align__(16) unsigned short sW[DOUT * DK];   // WcT [n][k] bf16, 16KB
  __shared__ float sB[DOUT];
  __shared__ __align__(16) unsigned short stag[8][K_ * DK]; // per-wave 32x64 bf16, 32KB
  const int tid = threadIdx.x, wid = tid >> 5, lane = tid & 31;
  for (int i = tid; i < DOUT * DK; i += 256) sW[i] = wct[i];
  for (int i = tid; i < DOUT; i += 256) sB[i] = bcomb[i];
  __syncthreads();

  const int gid = blockIdx.x * 8 + wid;   // global centroid row = b*2048+s
  const int b = gid >> 11;
  const int j = idx[gid * K_ + lane];     // this lane's neighbor point
  const float cx = newxyz[gid * 3 + 0], cy = newxyz[gid * 3 + 1], cz = newxyz[gid * 3 + 2];

  { // gather: lane writes feature row `lane` (centered xyz ++ points, pad->0)
    const float* xr = xyz + ((size_t)b * N_ + j) * 3;
    const float4* p4 = (const float4*)(points + ((size_t)b * N_ + j) * C_);
    float f[DK];
    f[0] = xr[0] - cx; f[1] = xr[1] - cy; f[2] = xr[2] - cz;
#pragma unroll
    for (int q = 0; q < 8; ++q) {
      float4 t = p4[q];
      f[3 + 4 * q + 0] = t.x; f[3 + 4 * q + 1] = t.y;
      f[3 + 4 * q + 2] = t.z; f[3 + 4 * q + 3] = t.w;
    }
#pragma unroll
    for (int c = DIN; c < DK; ++c) f[c] = 0.f;
    unsigned* rw = (unsigned*)&stag[wid][lane * DK];
#pragma unroll
    for (int c = 0; c < DK; c += 2) rw[c >> 1] = pack2(f[c], f[c + 1]);
  }
  __syncthreads();

  const int h = lane >> 4;       // ISA 16-bit layout half-select
  const int m = lane & 15;
  // A fragments: lane holds row (mt*16+m); elems 0..7 at k0=ks*32+h*8, 8..15 at +16
  FragU a[2][2];
#pragma unroll
  for (int mt = 0; mt < 2; ++mt)
#pragma unroll
    for (int ks = 0; ks < 2; ++ks) {
      const unsigned short* rp = &stag[wid][(mt * 16 + m) * DK];
      int k0 = ks * 32 + h * 8;
      a[mt][ks].f[0] = *(const float4*)(rp + k0);
      a[mt][ks].f[1] = *(const float4*)(rp + k0 + 16);
    }

  float* orow = newpts + (size_t)gid * DOUT;
#pragma unroll
  for (int nt = 0; nt < 8; ++nt) {
    // B fragment: lane holds col n=nt*16+m; elems 0..7 at k0=ks*32+h*8, 8..15 at +16
    FragU bf[2];
    const unsigned short* wp = &sW[(nt * 16 + m) * DK];
#pragma unroll
    for (int ks = 0; ks < 2; ++ks) {
      int k0 = ks * 32 + h * 8;
      bf[ks].f[0] = *(const float4*)(wp + k0);
      bf[ks].f[1] = *(const float4*)(wp + k0 + 16);
    }
    v8f acc0 = {0, 0, 0, 0, 0, 0, 0, 0};
    v8f acc1 = {0, 0, 0, 0, 0, 0, 0, 0};
#pragma unroll
    for (int ks = 0; ks < 2; ++ks) {
      acc0 = __builtin_amdgcn_wmma_f32_16x16x32_bf16(false, a[0][ks].v, false, bf[ks].v,
                                                     (short)0, acc0, false, false);
      acc1 = __builtin_amdgcn_wmma_f32_16x16x32_bf16(false, a[1][ks].v, false, bf[ks].v,
                                                     (short)0, acc1, false, false);
    }
    // maxpool over the 32 neighbor rows: 8 VGPRs x 2 M-tiles per lane + xor16
    float mx = -3.4e38f;
#pragma unroll
    for (int e = 0; e < 8; ++e) { mx = fmaxf(mx, acc0[e]); mx = fmaxf(mx, acc1[e]); }
    mx = fmaxf(mx, __shfl_xor(mx, 16, 32));
    if (lane < 16) {
      int ch = nt * 16 + lane;
      orow[ch] = mx + sB[ch];    // bias after max: valid (per-channel constant)
    }
  }
}

// ---------------------------------------------------------------------------
extern "C" void kernel_launch(void* const* d_in, const int* in_sizes, int n_in,
                              void* d_out, int out_size, void* d_ws, size_t ws_size,
                              hipStream_t stream) {
  (void)in_sizes; (void)n_in; (void)out_size; (void)ws_size;
  const float* xyz    = (const float*)d_in[0];
  const float* points = (const float*)d_in[1];
  const float* w0 = (const float*)d_in[2];
  const float* b0 = (const float*)d_in[3];
  const float* w1 = (const float*)d_in[4];
  const float* b1 = (const float*)d_in[5];
  const float* w2 = (const float*)d_in[6];
  const float* b2 = (const float*)d_in[7];

  float* out    = (float*)d_out;
  float* newxyz = out;                                       // B*S*3
  float* newpts = out + (size_t)B_ * S_ * 3;                 // B*S*128
  int*   idxout = (int*)(out + (size_t)B_ * S_ * (3 + DOUT)); // B*S*32 (int bits)

  unsigned short* wct = (unsigned short*)d_ws;               // 64*128 bf16
  float* bc = (float*)((char*)d_ws + (size_t)DOUT * DK * sizeof(unsigned short));

  compose_kernel<<<1, 256, 0, stream>>>(w0, b0, w1, b1, w2, b2, wct, bc);

  size_t fps_lds = (size_t)N_ * 3 * sizeof(float) + 32 * sizeof(float) + 33 * sizeof(int);
  fps_kernel<<<B_, 1024, fps_lds, stream>>>(xyz, newxyz);

  size_t ball_lds = (size_t)N_ * 3 * sizeof(float);
  ball_kernel<<<(B_ * S_) / 8, 256, ball_lds, stream>>>(xyz, newxyz, idxout);

  mlp_kernel<<<(B_ * S_) / 8, 256, 0, stream>>>(xyz, points, wct, bc, newxyz, idxout, newpts);
}
